// config1_2319282339850
// MI455X (gfx1250) — matrix operand order repaired and checked by reading, not compiled, not run on hardware
//
#include <hip/hip_runtime.h>

// GNN: 3x GraphConv + 3x SAGEConv + JK(max) x2 + MLP head, on gfx1250 (MI455X).
// GEMMs: v_wmma_f32_16x16x32_bf16 (fp32 accumulate), A staged in LDS.
// Scatter-max: branchless order-preserving uint mapping + global_atomic_max_u32.

#define NN 50000
#define NE 800000
#define HID 128
#define LDS_PAD 8   // ushorts of row padding => conflict-free ds_load_b128

typedef __attribute__((ext_vector_type(16))) __bf16 v16bf;
typedef __attribute__((ext_vector_type(8)))  float  v8f;

__device__ __forceinline__ unsigned short f2bf(float f) {
    unsigned int u = __float_as_uint(f);
    unsigned int r = u + 0x7FFFu + ((u >> 16) & 1u);   // round-to-nearest-even
    return (unsigned short)(r >> 16);
}

// Order-preserving float -> uint map: float compare == unsigned compare.
__device__ __forceinline__ unsigned int fmapk(float f) {
    unsigned int u = __float_as_uint(f);
    return (u & 0x80000000u) ? ~u : (u | 0x80000000u);
}
__device__ __forceinline__ float funmapk(unsigned int u) {
    return __uint_as_float((u & 0x80000000u) ? (u ^ 0x80000000u) : ~u);
}

// ---- prep: W[K,128] fp32 -> Wt[128,K] bf16 (output-major => contiguous K) ----
__global__ void k_wprep(const float* __restrict__ W, unsigned short* __restrict__ Wt,
                        int K, int Ncol) {
    int idx = blockIdx.x * blockDim.x + threadIdx.x;
    if (idx >= K * Ncol) return;
    int k = idx / Ncol, n = idx % Ncol;
    Wt[(long long)n * K + k] = f2bf(W[(long long)k * Ncol + n]);
}

__global__ void k_f2bf(const float* __restrict__ s, unsigned short* __restrict__ d, long long n) {
    long long i = (long long)blockIdx.x * blockDim.x + threadIdx.x;
    if (i < n) d[i] = f2bf(s[i]);
}

// ---- edge-parallel scatter-max in mapped-uint domain (agg zero-initialized) ----
__global__ void k_scatter_max(const float* __restrict__ h, const long long* __restrict__ ei,
                              const float* __restrict__ ew, unsigned int* __restrict__ agg,
                              int D, int gshift) {
    long long tid = (long long)blockIdx.x * blockDim.x + threadIdx.x;
    int groups = D >> 2;
    long long total = (long long)NE * groups;
    if (tid >= total) return;
    long long e = tid >> gshift;
    int g = (int)(tid & (groups - 1));
    int s = (int)ei[e];
    int d = (int)ei[NE + e];
    float w = 1.0f / ew[e];
    const float4 v = *(const float4*)(h + (long long)s * D + g * 4);
    unsigned int* a = agg + (long long)d * D + g * 4;
    atomicMax(a + 0, fmapk(v.x * w));
    atomicMax(a + 1, fmapk(v.y * w));
    atomicMax(a + 2, fmapk(v.z * w));
    atomicMax(a + 3, fmapk(v.w * w));
}

// ---- decode mapped uint (0 == isolated node -> 0.0f) and convert to bf16 ----
__global__ void k_fixup_cvt(const unsigned int* __restrict__ agg,
                            unsigned short* __restrict__ aggB, long long n) {
    long long i = (long long)blockIdx.x * blockDim.x + threadIdx.x;
    if (i >= n) return;
    unsigned int u = agg[i];
    float v = (u == 0u) ? 0.0f : funmapk(u);
    aggB[i] = f2bf(v);
}

// ---- WMMA operand loaders (ISA 7.12.2 layouts, wave32) ----
// A 16x32 bf16 from LDS (row stride = K+LDS_PAD ushorts):
// lanes 0-15 row m, K {0..7,16..23}; lanes 16-31 row m, K {8..15,24..31}
__device__ __forceinline__ v16bf lds_A16x32(const unsigned short* s, int strideU,
                                            int k0, int lane) {
    int m = lane & 15, half = lane >> 4;
    const unsigned short* p = s + m * strideU + k0 + half * 8;
    union { uint4 q[2]; v16bf v; } u;
    u.q[0] = *(const uint4*)p;          // K = k0+half*8    .. +7
    u.q[1] = *(const uint4*)(p + 16);   // K = k0+16+half*8 .. +7
    return u.v;
}
// B 32x16 bf16 from global: lane n = column, lanes 0-15 K {0..7,16..23}, lanes 16-31 K {8..15,24..31}
__device__ __forceinline__ v16bf load_B32x16(const unsigned short* base, int col0,
                                             int K, int k0, int lane) {
    int n = lane & 15, half = lane >> 4;
    const unsigned short* p = base + (long long)(col0 + n) * K + k0 + half * 8;
    union { uint4 q[2]; v16bf v; } u;
    u.q[0] = *(const uint4*)p;          // K = k0+half*8    .. +7
    u.q[1] = *(const uint4*)(p + 16);   // K = k0+16+half*8 .. +7
    return u.v;
}

// ---- fused dual GEMM + bias + PReLU + JK running max + bf16 copy ----
// out = prelu(A1*B1t^T + A2*B2t^T + bias).  Block = 256 thr = 8 waves = 16x128 tile.
// Both 16xK A tiles staged in LDS once per block (8x reuse across waves).
__global__ void k_gemm_dual(const unsigned short* __restrict__ A1,
                            const unsigned short* __restrict__ A2,
                            const unsigned short* __restrict__ B1t,
                            const unsigned short* __restrict__ B2t,
                            const float* __restrict__ bias,
                            const float* __restrict__ slope,
                            float* __restrict__ outF, unsigned short* __restrict__ outB,
                            float* __restrict__ jk, int jk_init, int K) {
    __shared__ __align__(16) unsigned short sA1[16 * (HID + LDS_PAD)];
    __shared__ __align__(16) unsigned short sA2[16 * (HID + LDS_PAD)];

    int lane = threadIdx.x & 31;
    int wave = threadIdx.x >> 5;
    int col0 = wave * 16;
    long long row0 = (long long)blockIdx.x * 16;
    int strideU = K + LDS_PAD;

    // cooperative stage: 16 rows x K bf16 per matrix, uint4 (8 ushort) granules
    {
        int t = threadIdx.x;
        int qpr = K >> 3;                 // uint4 quads per row
        if (t < 16 * qpr) {
            int r = t / qpr, q = t - r * qpr;
            const uint4* g1 = (const uint4*)(A1 + (row0 + r) * (long long)K);
            const uint4* g2 = (const uint4*)(A2 + (row0 + r) * (long long)K);
            ((uint4*)(sA1 + r * strideU))[q] = g1[q];
            ((uint4*)(sA2 + r * strideU))[q] = g2[q];
        }
    }
    __syncthreads();

    v8f c = {0.f, 0.f, 0.f, 0.f, 0.f, 0.f, 0.f, 0.f};
    for (int k0 = 0; k0 < K; k0 += 32) {
        v16bf a = lds_A16x32(sA1, strideU, k0, lane);
        v16bf b = load_B32x16(B1t, col0, K, k0, lane);
        c = __builtin_amdgcn_wmma_f32_16x16x32_bf16(false, a, false, b, (short)0, c, false, false);
    }
    for (int k0 = 0; k0 < K; k0 += 32) {
        v16bf a = lds_A16x32(sA2, strideU, k0, lane);
        v16bf b = load_B32x16(B2t, col0, K, k0, lane);
        c = __builtin_amdgcn_wmma_f32_16x16x32_bf16(false, a, false, b, (short)0, c, false, false);
    }

    int n = lane & 15, half = lane >> 4;
    int col = col0 + n;
    float bcol = bias[col], acol = slope[col];
#pragma unroll
    for (int v = 0; v < 8; v++) {
        long long m = row0 + half * 8 + v;      // C/D layout: VGPR v, halves at m, m+8
        float val = c[v] + bcol;
        val = (val >= 0.0f) ? val : acol * val; // PReLU
        long long idx = m * HID + col;
        outF[idx] = val;
        outB[idx] = f2bf(val);
        if (jk_init) jk[idx] = val;
        else         jk[idx] = fmaxf(jk[idx], val);
    }
}

// ---- head: 128 -> 63 (PReLU), 63 -> 3 ----
__global__ void k_head1(const float* __restrict__ h, const float* __restrict__ W,
                        const float* __restrict__ b, const float* __restrict__ a,
                        float* __restrict__ out) {
    long long idx = (long long)blockIdx.x * blockDim.x + threadIdx.x;
    if (idx >= (long long)NN * 63) return;
    int n = (int)(idx / 63), cc = (int)(idx % 63);
    const float* hp = h + (long long)n * HID;
    float acc = b[cc];
    for (int k = 0; k < HID; k++) acc = fmaf(hp[k], W[k * 63 + cc], acc);
    out[idx] = (acc >= 0.0f) ? acc : a[cc] * acc;
}

__global__ void k_head2(const float* __restrict__ h, const float* __restrict__ W,
                        const float* __restrict__ b, float* __restrict__ out) {
    long long idx = (long long)blockIdx.x * blockDim.x + threadIdx.x;
    if (idx >= (long long)NN * 3) return;
    int n = (int)(idx / 3), cc = (int)(idx % 3);
    const float* hp = h + (long long)n * 63;
    float acc = b[cc];
    for (int k = 0; k < 63; k++) acc = fmaf(hp[k], W[k * 3 + cc], acc);
    out[idx] = acc;
}

extern "C" void kernel_launch(void* const* d_in, const int* in_sizes, int n_in,
                              void* d_out, int out_size, void* d_ws, size_t ws_size,
                              hipStream_t stream) {
    (void)in_sizes; (void)n_in; (void)out_size; (void)ws_size;
    const float*     x  = (const float*)d_in[0];
    const long long* ei = (const long long*)d_in[1];   // [2, NE] int64
    const float*     ew = (const float*)d_in[2];
    float* out = (float*)d_out;

    // workspace carve-up (256B aligned)
    char* ws = (char*)d_ws;
    size_t off = 0;
    auto take = [&](size_t bytes) -> char* {
        char* p = ws + off;
        off = (off + bytes + 255) & ~(size_t)255;
        return p;
    };
    unsigned int*   aggU = (unsigned int*)take((size_t)NN * HID * 4);
    unsigned short* aggB = (unsigned short*)take((size_t)NN * HID * 2);
    float*          hF[2];  unsigned short* hB[2];
    hF[0] = (float*)take((size_t)NN * HID * 4);
    hF[1] = (float*)take((size_t)NN * HID * 4);
    hB[0] = (unsigned short*)take((size_t)NN * HID * 2);
    hB[1] = (unsigned short*)take((size_t)NN * HID * 2);
    float*          jkF = (float*)take((size_t)NN * HID * 4);
    unsigned short* jkB = (unsigned short*)take((size_t)NN * HID * 2);
    unsigned short* xB  = (unsigned short*)take((size_t)NN * 64 * 2);
    float*          t63 = (float*)take((size_t)NN * 63 * 4);
    unsigned short* wt[12];
    for (int i = 0; i < 12; i++) wt[i] = (unsigned short*)take((size_t)HID * HID * 2);

    const int B = 256;

    // weight prep: layer l weights at d_in[3+3l] (rel/l), [5+3l] (root/r); bias [4+3l]
    for (int l = 0; l < 6; l++) {
        int K = (l == 0) ? 64 : HID;
        int nEl = K * HID;
        k_wprep<<<(nEl + B - 1) / B, B, 0, stream>>>((const float*)d_in[3 + 3 * l], wt[2 * l],     K, HID);
        k_wprep<<<(nEl + B - 1) / B, B, 0, stream>>>((const float*)d_in[5 + 3 * l], wt[2 * l + 1], K, HID);
    }
    {
        long long n = (long long)NN * 64;
        k_f2bf<<<(unsigned)((n + B - 1) / B), B, 0, stream>>>(x, xB, n);
    }

    const float*          curF = x;
    const unsigned short* curB = xB;
    int pp = 0;
    for (int l = 0; l < 6; l++) {
        int K = (l == 0) ? 64 : HID;
        long long nAgg = (long long)NN * K;
        int gshift = (K == 64) ? 4 : 5;                     // log2(K/4)
        long long nEdgeThreads = (long long)NE * (K >> 2);

        hipMemsetAsync(aggU, 0, (size_t)nAgg * 4, stream);  // 0 == mapped "empty"
        k_scatter_max<<<(unsigned)((nEdgeThreads + B - 1) / B), B, 0, stream>>>(curF, ei, ew, aggU, K, gshift);
        k_fixup_cvt<<<(unsigned)((nAgg + B - 1) / B), B, 0, stream>>>(aggU, aggB, nAgg);

        k_gemm_dual<<<NN / 16, 256, 0, stream>>>(aggB, curB, wt[2 * l], wt[2 * l + 1],
                                                 (const float*)d_in[4 + 3 * l],   // bias
                                                 (const float*)d_in[25 + l],      // PReLU slope a_{l+1}
                                                 hF[pp], hB[pp], jkF,
                                                 (l % 3 == 0) ? 1 : 0, K);
        curF = hF[pp];
        curB = hB[pp];
        pp ^= 1;

        if (l == 2) {   // JumpingKnowledge(max) over layers 1-3 -> input of SAGE stack
            long long n = (long long)NN * HID;
            k_f2bf<<<(unsigned)((n + B - 1) / B), B, 0, stream>>>(jkF, jkB, n);
            curF = jkF;
            curB = jkB;
        }
    }
    // jkF now holds JK(max) over layers 4-6 -> head
    {
        long long n1 = (long long)NN * 63;
        k_head1<<<(unsigned)((n1 + B - 1) / B), B, 0, stream>>>(jkF, (const float*)d_in[21],
                                                                (const float*)d_in[22],
                                                                (const float*)d_in[31], t63);
        long long n2 = (long long)NN * 3;
        k_head2<<<(unsigned)((n2 + B - 1) / B), B, 0, stream>>>(t63, (const float*)d_in[23],
                                                                (const float*)d_in[24], out);
    }
}
